// GATTwoLayer_2001454760656
// MI455X (gfx1250) — compile-verified, matrix operand order checked by reading, not run on hardware
//
#include <hip/hip_runtime.h>
#include <math.h>
#include <stdint.h>

// CDNA5 wave32 WMMA types
typedef __attribute__((ext_vector_type(16))) _Float16 v16h;
typedef __attribute__((ext_vector_type(8)))  float    v8f;

#define F_IN   128
#define HEADS  8
#define HID    32
#define C1     (HEADS * HID)   // 256
#define NCLS   16
#define NEG_SLOPE 0.2f

// K index pattern for 16-bit A/B operands of v_wmma_*_16x16x32 (wave32):
// lanes 0-15 hold K={0..7,16..23}, lanes 16-31 hold K={8..15,24..31}
__device__ __forceinline__ int kpat(int half, int i) {
    return (i < 8) ? (half * 8 + i) : (16 + half * 8 + (i - 8));
}

// float atomic max via int/uint punning (correct for mixed signs, init to -inf)
__device__ __forceinline__ void atomicMaxF(float* addr, float v) {
    if (v >= 0.f) atomicMax((int*)addr, __float_as_int(v));
    else          atomicMin((unsigned int*)addr, __float_as_uint(v));
}

// pack one K-step of an A-row (contiguous in memory) into WMMA layout
__device__ __forceinline__ v16h load_a_row(const float4* rowv, int ks, int half, bool relu) {
    float4 a0 = rowv[(ks + half * 8) >> 2];
    float4 a1 = rowv[((ks + half * 8) >> 2) + 1];
    float4 a2 = rowv[(ks + 16 + half * 8) >> 2];
    float4 a3 = rowv[((ks + 16 + half * 8) >> 2) + 1];
    if (relu) {
        a0.x = fmaxf(a0.x, 0.f); a0.y = fmaxf(a0.y, 0.f); a0.z = fmaxf(a0.z, 0.f); a0.w = fmaxf(a0.w, 0.f);
        a1.x = fmaxf(a1.x, 0.f); a1.y = fmaxf(a1.y, 0.f); a1.z = fmaxf(a1.z, 0.f); a1.w = fmaxf(a1.w, 0.f);
        a2.x = fmaxf(a2.x, 0.f); a2.y = fmaxf(a2.y, 0.f); a2.z = fmaxf(a2.z, 0.f); a2.w = fmaxf(a2.w, 0.f);
        a3.x = fmaxf(a3.x, 0.f); a3.y = fmaxf(a3.y, 0.f); a3.z = fmaxf(a3.z, 0.f); a3.w = fmaxf(a3.w, 0.f);
    }
    v16h a;
    a[0]  = (_Float16)a0.x; a[1]  = (_Float16)a0.y; a[2]  = (_Float16)a0.z; a[3]  = (_Float16)a0.w;
    a[4]  = (_Float16)a1.x; a[5]  = (_Float16)a1.y; a[6]  = (_Float16)a1.z; a[7]  = (_Float16)a1.w;
    a[8]  = (_Float16)a2.x; a[9]  = (_Float16)a2.y; a[10] = (_Float16)a2.z; a[11] = (_Float16)a2.w;
    a[12] = (_Float16)a3.x; a[13] = (_Float16)a3.y; a[14] = (_Float16)a3.z; a[15] = (_Float16)a3.w;
    return a;
}

// ---------------- Layer 1 GEMM: h1[N,256] = x[N,128] @ W1[128,256] ----------
__global__ void gemm1_wmma(const float* __restrict__ x, const float* __restrict__ W,
                           float* __restrict__ h1, int Nn) {
    int wave = blockIdx.x * (blockDim.x >> 5) + (threadIdx.x >> 5);
    int lane = threadIdx.x & 31;
    const int ntiles = C1 / 16;              // 16 column tiles
    int mtile = wave / ntiles;
    int ntile = wave % ntiles;
    int mbase = mtile * 16;
    if (mbase >= Nn) return;
    int m    = lane & 15;
    int half = lane >> 4;
    bool full = (mbase + 16 <= Nn);
    int row  = mbase + m;                    // A row for this lane
    if (row >= Nn) row = Nn - 1;             // clamp tail loads (masked at store)
    int col  = ntile * 16 + m;               // B column for this lane
    const float4* rowv = (const float4*)(x + (size_t)row * F_IN);
    v8f acc = {0.f, 0.f, 0.f, 0.f, 0.f, 0.f, 0.f, 0.f};
#pragma unroll
    for (int ks = 0; ks < F_IN; ks += 32) {
        v16h a = load_a_row(rowv, ks, half, false);
        v16h b;
#pragma unroll
        for (int i = 0; i < 16; ++i) {
            int k = ks + kpat(half, i);
            b[i] = (_Float16)W[(size_t)k * C1 + col];
        }
        acc = __builtin_amdgcn_wmma_f32_16x16x32_f16(false, a, false, b,
                                                     (short)0, acc, false, false);
    }
    float* outp = h1 + (size_t)(mbase + 8 * half) * C1 + ntile * 16 + (lane & 15);
    if (full) {
#pragma unroll
        for (int r = 0; r < 8; ++r)
            outp[(size_t)r * C1] = acc[r];
    } else {
#pragma unroll
        for (int r = 0; r < 8; ++r) {
            int orow = mbase + r + 8 * half;
            if (orow < Nn) h1[(size_t)orow * C1 + ntile * 16 + (lane & 15)] = acc[r];
        }
    }
}

// -------- Layer 2 GEMM: h2[N,16] = relu(out1[N,256]) @ W2[256,16] -----------
__global__ void gemm2_wmma(const float* __restrict__ out1, const float* __restrict__ W2,
                           float* __restrict__ h2, int Nn) {
    int wave = blockIdx.x * (blockDim.x >> 5) + (threadIdx.x >> 5);
    int lane = threadIdx.x & 31;
    int mbase = wave * 16;
    if (mbase >= Nn) return;
    int m    = lane & 15;
    int half = lane >> 4;
    bool full = (mbase + 16 <= Nn);
    int row  = mbase + m;
    if (row >= Nn) row = Nn - 1;
    const float4* rowv = (const float4*)(out1 + (size_t)row * C1);
    v8f acc = {0.f, 0.f, 0.f, 0.f, 0.f, 0.f, 0.f, 0.f};
#pragma unroll
    for (int ks = 0; ks < C1; ks += 32) {
        v16h a = load_a_row(rowv, ks, half, true);   // fused ReLU
        v16h b;
#pragma unroll
        for (int i = 0; i < 16; ++i) {
            int k = ks + kpat(half, i);
            b[i] = (_Float16)W2[(size_t)k * NCLS + m];
        }
        acc = __builtin_amdgcn_wmma_f32_16x16x32_f16(false, a, false, b,
                                                     (short)0, acc, false, false);
    }
    float* outp = h2 + (size_t)(mbase + 8 * half) * NCLS + (lane & 15);
    if (full) {
#pragma unroll
        for (int r = 0; r < 8; ++r)
            outp[(size_t)r * NCLS] = acc[r];
    } else {
#pragma unroll
        for (int r = 0; r < 8; ++r) {
            int orow = mbase + r + 8 * half;
            if (orow < Nn) h2[(size_t)orow * NCLS + (lane & 15)] = acc[r];
        }
    }
}

// ---------------- attention score reductions --------------------------------
__global__ void attn1_kernel(const float* __restrict__ h1,
                             const float* __restrict__ att_s, const float* __restrict__ att_d,
                             float* __restrict__ as1, float* __restrict__ ad1, int Nn) {
    int t = blockIdx.x * blockDim.x + threadIdx.x;
    if (t >= Nn * HEADS) return;
    int n = t / HEADS, h = t % HEADS;
    const float* hv = h1 + (size_t)n * C1 + h * HID;
    const float* sv = att_s + h * HID;
    const float* dv = att_d + h * HID;
    float ss = 0.f, sd = 0.f;
#pragma unroll
    for (int c = 0; c < HID; ++c) { float v = hv[c]; ss += v * sv[c]; sd += v * dv[c]; }
    as1[t] = ss; ad1[t] = sd;
}

__global__ void attn2_kernel(const float* __restrict__ h2,
                             const float* __restrict__ att_s, const float* __restrict__ att_d,
                             float* __restrict__ as2, float* __restrict__ ad2, int Nn) {
    int n = blockIdx.x * blockDim.x + threadIdx.x;
    if (n >= Nn) return;
    float ss = 0.f, sd = 0.f;
#pragma unroll
    for (int c = 0; c < NCLS; ++c) { float v = h2[(size_t)n * NCLS + c]; ss += v * att_s[c]; sd += v * att_d[c]; }
    as2[n] = ss; ad2[n] = sd;
}

// ---------------- init kernels ----------------------------------------------
__global__ void init1_kernel(const float* __restrict__ b1, float* __restrict__ out1,
                             float* __restrict__ m1, float* __restrict__ d1, int Nn) {
    int t = blockIdx.x * blockDim.x + threadIdx.x;
    if (t < Nn * C1)    out1[t] = b1[t & (C1 - 1)];
    if (t < Nn * HEADS) { m1[t] = -__builtin_inff(); d1[t] = 0.f; }
}

__global__ void init2_kernel(const float* __restrict__ b2, float* __restrict__ out,
                             float* __restrict__ m2, float* __restrict__ d2, int Nn) {
    int t = blockIdx.x * blockDim.x + threadIdx.x;
    if (t < Nn * NCLS) out[t] = b2[t & (NCLS - 1)];
    if (t < Nn)        { m2[t] = -__builtin_inff(); d2[t] = 0.f; }
}

// ---------------- edge passes, layer 1 --------------------------------------
__global__ void edge_max1(const long long* __restrict__ ei, int E, int Nn,
                          const float* __restrict__ as1, const float* __restrict__ ad1,
                          float* __restrict__ m1) {
    int t = blockIdx.x * blockDim.x + threadIdx.x;
    int Etot = E + Nn;
    if (t >= Etot) return;
    int s = (t < E) ? (int)ei[t]     : (t - E);   // self-loops synthesized
    int d = (t < E) ? (int)ei[E + t] : (t - E);
#pragma unroll
    for (int h = 0; h < HEADS; ++h) {
        float v = as1[s * HEADS + h] + ad1[d * HEADS + h];
        v = (v > 0.f) ? v : NEG_SLOPE * v;
        atomicMaxF(&m1[d * HEADS + h], v);
    }
}

__global__ void edge_sum1(const long long* __restrict__ ei, int E, int Nn,
                          const float* __restrict__ as1, const float* __restrict__ ad1,
                          const float* __restrict__ m1, float* __restrict__ d1) {
    int t = blockIdx.x * blockDim.x + threadIdx.x;
    int Etot = E + Nn;
    if (t >= Etot) return;
    int s = (t < E) ? (int)ei[t]     : (t - E);
    int d = (t < E) ? (int)ei[E + t] : (t - E);
#pragma unroll
    for (int h = 0; h < HEADS; ++h) {
        float v = as1[s * HEADS + h] + ad1[d * HEADS + h];
        v = (v > 0.f) ? v : NEG_SLOPE * v;
        atomicAdd(&d1[d * HEADS + h], __expf(v - m1[d * HEADS + h]));
    }
}

// wave-per-edge aggregation: lane = channel, k loop = head, alpha via shfl
__global__ void agg1(const long long* __restrict__ ei, int E, int Nn,
                     const float* __restrict__ as1, const float* __restrict__ ad1,
                     const float* __restrict__ m1, const float* __restrict__ d1,
                     const float* __restrict__ h1, float* __restrict__ out1) {
    int wave = blockIdx.x * (blockDim.x >> 5) + (threadIdx.x >> 5);
    int lane = threadIdx.x & 31;
    int Etot = E + Nn;
    if (wave >= Etot) return;
    int s = (wave < E) ? (int)ei[wave]     : (wave - E);
    int d = (wave < E) ? (int)ei[E + wave] : (wave - E);
    int hh = lane & 7;
    float v = as1[s * HEADS + hh] + ad1[d * HEADS + hh];
    v = (v > 0.f) ? v : NEG_SLOPE * v;
    float aval = __expf(v - m1[d * HEADS + hh]) / (d1[d * HEADS + hh] + 1e-16f);
#pragma unroll
    for (int k = 0; k < HEADS; ++k) {
        float alpha = __shfl(aval, k, 32);    // wave32 broadcast of head-k weight
        float val = alpha * h1[(size_t)s * C1 + k * HID + lane];
        atomicAdd(&out1[(size_t)d * C1 + k * HID + lane], val);
    }
}

// ---------------- edge passes, layer 2 (1 head, 16 channels) ----------------
__global__ void edge_max2(const long long* __restrict__ ei, int E, int Nn,
                          const float* __restrict__ as2, const float* __restrict__ ad2,
                          float* __restrict__ m2) {
    int t = blockIdx.x * blockDim.x + threadIdx.x;
    int Etot = E + Nn;
    if (t >= Etot) return;
    int s = (t < E) ? (int)ei[t]     : (t - E);
    int d = (t < E) ? (int)ei[E + t] : (t - E);
    float v = as2[s] + ad2[d];
    v = (v > 0.f) ? v : NEG_SLOPE * v;
    atomicMaxF(&m2[d], v);
}

__global__ void edge_sum2(const long long* __restrict__ ei, int E, int Nn,
                          const float* __restrict__ as2, const float* __restrict__ ad2,
                          const float* __restrict__ m2, float* __restrict__ d2) {
    int t = blockIdx.x * blockDim.x + threadIdx.x;
    int Etot = E + Nn;
    if (t >= Etot) return;
    int s = (t < E) ? (int)ei[t]     : (t - E);
    int d = (t < E) ? (int)ei[E + t] : (t - E);
    float v = as2[s] + ad2[d];
    v = (v > 0.f) ? v : NEG_SLOPE * v;
    atomicAdd(&d2[d], __expf(v - m2[d]));
}

__global__ void agg2(const long long* __restrict__ ei, int E, int Nn,
                     const float* __restrict__ as2, const float* __restrict__ ad2,
                     const float* __restrict__ m2, const float* __restrict__ d2,
                     const float* __restrict__ h2, float* __restrict__ out) {
    int t = blockIdx.x * blockDim.x + threadIdx.x;
    int Etot = E + Nn;
    int e = t >> 4;
    int c = t & 15;
    if (e >= Etot) return;
    int s = (e < E) ? (int)ei[e]     : (e - E);
    int d = (e < E) ? (int)ei[E + e] : (e - E);
    float v = as2[s] + ad2[d];
    v = (v > 0.f) ? v : NEG_SLOPE * v;
    float alpha = __expf(v - m2[d]) / (d2[d] + 1e-16f);
    atomicAdd(&out[(size_t)d * NCLS + c], alpha * h2[(size_t)s * NCLS + c]);
}

// ---------------- launcher --------------------------------------------------
extern "C" void kernel_launch(void* const* d_in, const int* in_sizes, int n_in,
                              void* d_out, int out_size, void* d_ws, size_t ws_size,
                              hipStream_t stream) {
    const float*     x        = (const float*)d_in[0];
    const long long* ei       = (const long long*)d_in[1];   // int64 edge_index [2,E]
    const float*     W1       = (const float*)d_in[2];
    const float*     att_src1 = (const float*)d_in[3];
    const float*     att_dst1 = (const float*)d_in[4];
    const float*     b1       = (const float*)d_in[5];
    const float*     W2       = (const float*)d_in[6];
    const float*     att_src2 = (const float*)d_in[7];
    const float*     att_dst2 = (const float*)d_in[8];
    const float*     b2       = (const float*)d_in[9];
    float* out = (float*)d_out;

    int Nn = in_sizes[0] / F_IN;
    int E  = in_sizes[1] / 2;
    int Etot = E + Nn;

    // workspace carve-up (floats)
    float* ws  = (float*)d_ws;
    float* h1  = ws;  ws += (size_t)Nn * C1;
    float* out1= ws;  ws += (size_t)Nn * C1;
    float* as1 = ws;  ws += (size_t)Nn * HEADS;
    float* ad1 = ws;  ws += (size_t)Nn * HEADS;
    float* m1  = ws;  ws += (size_t)Nn * HEADS;
    float* d1  = ws;  ws += (size_t)Nn * HEADS;
    float* h2  = ws;  ws += (size_t)Nn * NCLS;
    float* as2 = ws;  ws += Nn;
    float* ad2 = ws;  ws += Nn;
    float* m2  = ws;  ws += Nn;
    float* d2  = ws;  ws += Nn;

    const int TB = 256;                       // 8 wave32 per block
    int mtiles = (Nn + 15) / 16;

    // ---- layer 1 ----
    {
        int waves  = mtiles * (C1 / 16);
        int blocks = (waves + 7) / 8;
        gemm1_wmma<<<blocks, TB, 0, stream>>>(x, W1, h1, Nn);
    }
    attn1_kernel<<<(Nn * HEADS + TB - 1) / TB, TB, 0, stream>>>(h1, att_src1, att_dst1, as1, ad1, Nn);
    init1_kernel<<<((size_t)Nn * C1 + TB - 1) / TB, TB, 0, stream>>>(b1, out1, m1, d1, Nn);
    edge_max1<<<(Etot + TB - 1) / TB, TB, 0, stream>>>(ei, E, Nn, as1, ad1, m1);
    edge_sum1<<<(Etot + TB - 1) / TB, TB, 0, stream>>>(ei, E, Nn, as1, ad1, m1, d1);
    agg1<<<(Etot + 7) / 8, TB, 0, stream>>>(ei, E, Nn, as1, ad1, m1, d1, h1, out1);

    // ---- layer 2 (ReLU fused into gemm2 A-load) ----
    {
        int blocks = (mtiles + 7) / 8;
        gemm2_wmma<<<blocks, TB, 0, stream>>>(out1, W2, h2, Nn);
    }
    attn2_kernel<<<(Nn + TB - 1) / TB, TB, 0, stream>>>(h2, att_src2, att_dst2, as2, ad2, Nn);
    init2_kernel<<<((size_t)Nn * NCLS + TB - 1) / TB, TB, 0, stream>>>(b2, out, m2, d2, Nn);
    edge_max2<<<(Etot + TB - 1) / TB, TB, 0, stream>>>(ei, E, Nn, as2, ad2, m2);
    edge_sum2<<<(Etot + TB - 1) / TB, TB, 0, stream>>>(ei, E, Nn, as2, ad2, m2, d2);
    agg2<<<((size_t)Etot * NCLS + TB - 1) / TB, TB, 0, stream>>>(ei, E, Nn, as2, ad2, m2, d2, h2, out);
}
